// STFormer_66589172957648
// MI455X (gfx1250) — matrix operand order repaired and checked by reading, not compiled, not run on hardware
//
#include <hip/hip_runtime.h>
#include <hip/hip_bf16.h>
#include <math.h>

// ---------------------------------------------------------------------------
// CDNA5 (gfx1250) STFormer forward.  All GEMMs via v_wmma_f32_16x16x32_bf16.
// Double-buffered LDS pipeline: global->reg loads for tile s+1 overlap the
// WMMA compute of tile s; one workgroup barrier per K-step.
// ---------------------------------------------------------------------------

typedef __attribute__((ext_vector_type(16))) __bf16 v16bf;
typedef __attribute__((ext_vector_type(8)))  float  v8f;

#define GBM 128
#define GBN 128
#define GBK 32
#define LSTR 40   // LDS row stride in bf16 elements (80 bytes, 16B aligned rows)

__device__ __forceinline__ float mishf(float x) {
    float sp = (x > 20.f) ? x : log1pf(expf(x));
    return x * tanhf(sp);
}

// ---------------------------- batched WMMA GEMM ----------------------------
struct GP {
    const __bf16* A; long long lda, sa0, sa1, sa2;
    const __bf16* B; long long ldb, sb0, sb1, sb2;
    float*        C; long long ldc, sc0, sc1, sc2;
    __bf16*     Cbf;          // optional bf16 dual write (same strides as C)
    const float* Cin;         // optional residual add
    const float* bias;        // [N]
    const float* bias2;       // row-broadcast table [*, b2ld]
    const int*   rowmap;      // row -> bias2 row
    int b2ld;
    int M, N, K;
    int D1, D2;               // batch z -> (z/(D1*D2), (z/D2)%D1, z%D2)
    int opB;                  // 0: B is KxN row-major; 1: B is NxK row-major
    int act;                  // 0 none, 1 mish
};

__global__ __launch_bounds__(256) void k_gemm(GP p) {
    __shared__ __align__(16) __bf16 As[2][GBM * LSTR];
    __shared__ __align__(16) __bf16 Bs[2][GBN * LSTR];

    const int tid  = threadIdx.x;
    const int lane = tid & 31;
    const int wave = tid >> 5;
    const int wm   = wave >> 1;      // 0..3
    const int wn   = wave & 1;       // 0..1
    const int lr   = lane & 15;
    const int hi   = lane >> 4;      // 0/1

    const int z   = blockIdx.z;
    const int d12 = p.D1 * p.D2;
    const int b0  = z / d12;
    const int r12 = z - b0 * d12;
    const int b1  = r12 / p.D2;
    const int b2  = r12 - b1 * p.D2;

    const __bf16* A  = p.A + (long long)b0 * p.sa0 + (long long)b1 * p.sa1 + (long long)b2 * p.sa2;
    const __bf16* Bm = p.B + (long long)b0 * p.sb0 + (long long)b1 * p.sb1 + (long long)b2 * p.sb2;
    const long long cOff = (long long)b0 * p.sc0 + (long long)b1 * p.sc1 + (long long)b2 * p.sc2;

    const int tM = blockIdx.y * GBM;
    const int tN = blockIdx.x * GBN;

    // per-thread global->LDS staging registers (one K-step ahead)
    uint4 ra[2], rb[2];

    auto loadA = [&](int k0) {
#pragma unroll
        for (int j = 0; j < 2; ++j) {
            int i = tid + 256 * j;            // 0..511 over 128x32/8 vectors
            int row = i >> 2, seg = i & 3;
            int gr = tM + row;
            uint4 v = make_uint4(0u, 0u, 0u, 0u);
            if (gr < p.M) v = *(const uint4*)(A + (long long)gr * p.lda + k0 + seg * 8);
            ra[j] = v;
        }
    };
    auto storeA = [&](int buf) {
#pragma unroll
        for (int j = 0; j < 2; ++j) {
            int i = tid + 256 * j;
            int row = i >> 2, seg = i & 3;
            *(uint4*)(&As[buf][row * LSTR + seg * 8]) = ra[j];
        }
    };
    auto loadB = [&](int k0) {
        if (p.opB) {                          // B stored NxK row-major (K-matrix)
#pragma unroll
            for (int j = 0; j < 2; ++j) {
                int i = tid + 256 * j;
                int row = i >> 2, seg = i & 3;
                int gn = tN + row;
                uint4 v = make_uint4(0u, 0u, 0u, 0u);
                if (gn < p.N) v = *(const uint4*)(Bm + (long long)gn * p.ldb + k0 + seg * 8);
                rb[j] = v;
            }
        } else {                              // B stored KxN row-major (weights, V)
#pragma unroll
            for (int j = 0; j < 2; ++j) {
                int i = tid + 256 * j;        // 0..511 over 32x128/8 vectors
                int k = i >> 4, n8 = (i & 15) * 8;
                int gn = tN + n8;             // N, ldb are multiples of 8 here
                uint4 v = make_uint4(0u, 0u, 0u, 0u);
                if (gn < p.N) v = *(const uint4*)(Bm + (long long)(k0 + k) * p.ldb + gn);
                rb[j] = v;
            }
        }
    };
    auto storeB = [&](int buf) {
        if (p.opB) {
#pragma unroll
            for (int j = 0; j < 2; ++j) {
                int i = tid + 256 * j;
                int row = i >> 2, seg = i & 3;
                *(uint4*)(&Bs[buf][row * LSTR + seg * 8]) = rb[j];
            }
        } else {                              // transpose into LDS as N x K
#pragma unroll
            for (int j = 0; j < 2; ++j) {
                int i = tid + 256 * j;
                int k = i >> 4, n8 = (i & 15) * 8;
                const __bf16* e = (const __bf16*)&rb[j];
#pragma unroll
                for (int t = 0; t < 8; ++t) Bs[buf][(n8 + t) * LSTR + k] = e[t];
            }
        }
    };

    v8f acc[2][4];
#pragma unroll
    for (int i = 0; i < 2; ++i)
#pragma unroll
        for (int j = 0; j < 4; ++j)
#pragma unroll
            for (int e = 0; e < 8; ++e) acc[i][j][e] = 0.f;

    union UA { v16bf v; uint4 q[2]; };

    // ---- pipelined main loop ----
    loadA(0); loadB(0);
    storeA(0); storeB(0);
    __syncthreads();

    const int nsteps = p.K / GBK;             // K padded to multiple of 32
    for (int s = 0; s < nsteps; ++s) {
        const int cur = s & 1;
        const bool hasNext = (s + 1 < nsteps);
        if (hasNext) {                        // issue next tile's global loads early
            loadA((s + 1) * GBK);
            loadB((s + 1) * GBK);
        }

        // fragments per documented 16-bit WMMA layouts
        UA ua[2], ub[4];
#pragma unroll
        for (int mf = 0; mf < 2; ++mf) {
            const __bf16* ap = &As[cur][(wm * 32 + mf * 16 + lr) * LSTR + hi * 8];
            ua[mf].q[0] = *(const uint4*)ap;          // K = kb .. kb+7
            ua[mf].q[1] = *(const uint4*)(ap + 16);   // K = kb+16 .. kb+23
        }
#pragma unroll
        for (int nf = 0; nf < 4; ++nf) {
            const __bf16* bp = &Bs[cur][(wn * 64 + nf * 16 + lr) * LSTR + hi * 8];
            ub[nf].q[0] = *(const uint4*)bp;          // K = kb .. kb+7
            ub[nf].q[1] = *(const uint4*)(bp + 16);   // K = kb+16 .. kb+23
        }
#pragma unroll
        for (int mf = 0; mf < 2; ++mf)
#pragma unroll
            for (int nf = 0; nf < 4; ++nf)
                acc[mf][nf] = __builtin_amdgcn_wmma_f32_16x16x32_bf16(
                    false, ua[mf].v, false, ub[nf].v, (short)0, acc[mf][nf], false, false);

        if (hasNext) {
            storeA(cur ^ 1);
            storeB(cur ^ 1);
            __syncthreads();                  // one barrier per K-step
        }
    }

    // ---- epilogue ----
    float*       Cp = p.C   ? p.C   + cOff : nullptr;
    __bf16*      Cb = p.Cbf ? p.Cbf + cOff : nullptr;
    const float* Ci = p.Cin ? p.Cin + cOff : nullptr;
#pragma unroll
    for (int mf = 0; mf < 2; ++mf) {
#pragma unroll
        for (int nf = 0; nf < 4; ++nf) {
            int gn = tN + wn * 64 + nf * 16 + lr;
            if (gn >= p.N) continue;
#pragma unroll
            for (int e = 0; e < 8; ++e) {
                int gm = tM + wm * 32 + mf * 16 + hi * 8 + e;
                if (gm >= p.M) continue;
                float v = acc[mf][nf][e];
                if (p.bias)  v += p.bias[gn];
                if (p.bias2) v += p.bias2[(long long)p.rowmap[gm] * p.b2ld + gn];
                if (p.act == 1) v = mishf(v);
                long long idx = (long long)gm * p.ldc + gn;
                if (Ci) v += Ci[idx];
                if (Cp) Cp[idx] = v;
                if (Cb) Cb[idx] = (__bf16)v;
            }
        }
    }
}

// ---------------------------- elementwise kernels ---------------------------
__global__ void k_cvt(const float* __restrict__ a, __bf16* __restrict__ o, long long n) {
    long long i = (long long)blockIdx.x * 256 + threadIdx.x;
    if (i < n) o[i] = (__bf16)a[i];
}

__global__ void k_add2(const float* __restrict__ a, const float* __restrict__ b,
                       float* __restrict__ o, long long n) {
    long long i = (long long)blockIdx.x * 256 + threadIdx.x;
    if (i < n) o[i] = a[i] + b[i];
}

// sin_emb -> bf16 (rows,128): [sin(x*f_j) | cos(x*f_j)], f_j = exp(-ln(1e4)/63 * j)
__global__ void k_sinemb(const int* __restrict__ tv, int useTime, int rows, __bf16* __restrict__ o) {
    int i = blockIdx.x * 256 + threadIdx.x;
    if (i >= rows * 64) return;
    int r = i >> 6, j = i & 63;
    float xv = useTime ? (float)tv[r] : (float)r;
    float f  = expf(-logf(10000.f) / 63.f * (float)j);
    float e  = xv * f;
    o[r * 128 + j]      = (__bf16)sinf(e);
    o[r * 128 + 64 + j] = (__bf16)cosf(e);
}

// te_mish[b*T+t, c] = mish(dte[b,c] + ste[t,c]);  B=8, T=128, C=128
__global__ void k_temish(const float* __restrict__ dte, const float* __restrict__ ste,
                         __bf16* __restrict__ o) {
    int i = blockIdx.x * 256 + threadIdx.x;   // 1024*128
    if (i >= 1024 * 128) return;
    int r = i >> 7, c = i & 127;
    int b = r >> 7, t = r & 127;
    o[i] = (__bf16)mishf(dte[b * 128 + c] + ste[t * 128 + c]);
}

// rowmap[r] = (r/dA)*mA + (r/dB)%mB
__global__ void k_rowmap(int* __restrict__ o, int n, int dA, int mA, int dB, int mB) {
    int r = blockIdx.x * 256 + threadIdx.x;
    if (r < n) o[r] = (r / dA) * mA + (r / dB) % mB;
}

// GroupNorm (reference semantics: stats per (batch, group) over ALL rows)
__global__ __launch_bounds__(256) void k_gn_stats(const float* __restrict__ x,
                                                  float* __restrict__ stat, int R) {
    int b = blockIdx.x >> 3, g = blockIdx.x & 7;
    const float* xp = x + (long long)b * R * 128 + g * 16;
    float s = 0.f, ss = 0.f;
    for (int i = threadIdx.x; i < R * 16; i += 256) {
        int r = i >> 4, c = i & 15;
        float v = xp[(long long)r * 128 + c];
        s += v; ss += v * v;
    }
    __shared__ float ls[256], lq[256];
    ls[threadIdx.x] = s; lq[threadIdx.x] = ss;
    __syncthreads();
    for (int o2 = 128; o2 > 0; o2 >>= 1) {
        if (threadIdx.x < o2) { ls[threadIdx.x] += ls[threadIdx.x + o2]; lq[threadIdx.x] += lq[threadIdx.x + o2]; }
        __syncthreads();
    }
    if (threadIdx.x == 0) {
        float n  = (float)R * 16.f;
        float m  = ls[0] / n;
        float va = lq[0] / n - m * m;
        stat[blockIdx.x * 2]     = m;
        stat[blockIdx.x * 2 + 1] = rsqrtf(va + 1e-5f);
    }
}

__global__ void k_gn_apply(const float* __restrict__ x, const float* __restrict__ stat,
                           const float* __restrict__ gm, const float* __restrict__ bt,
                           float* __restrict__ of, __bf16* __restrict__ ob, int R, long long n) {
    long long i = (long long)blockIdx.x * 256 + threadIdx.x;
    if (i >= n) return;
    int c = (int)(i & 127);
    int b = (int)(i / ((long long)R * 128));
    int g = c >> 4;
    float m  = stat[(b * 8 + g) * 2];
    float rs = stat[(b * 8 + g) * 2 + 1];
    float v  = (x[i] - m) * rs * gm[c] + bt[c];
    if (of) of[i] = v;
    ob[i] = (__bf16)v;
}

// row softmax with scale folded; pads (c>=ncol) emit zero probs (bf16)
__global__ __launch_bounds__(256) void k_softmax(const float* __restrict__ S, __bf16* __restrict__ P,
                                                 long long rows, int ncol, int npad, float scale) {
    long long row = (long long)blockIdx.x * 8 + (threadIdx.x >> 5);
    int lane = threadIdx.x & 31;
    if (row >= rows) return;
    const float* sr = S + row * npad;
    __bf16*      pr = P + row * npad;
    int nv = npad >> 5;   // 1..4
    float v[4];
    float m = -3.0e38f;
    for (int i = 0; i < nv; ++i) {
        int c = lane + (i << 5);
        float t = (c < ncol) ? sr[c] * scale : -3.0e38f;
        v[i] = t; m = fmaxf(m, t);
    }
    for (int o = 16; o; o >>= 1) m = fmaxf(m, __shfl_xor(m, o, 32));
    float s = 0.f;
    for (int i = 0; i < nv; ++i) {
        int c = lane + (i << 5);
        float e = (c < ncol) ? expf(v[i] - m) : 0.f;
        v[i] = e; s += e;
    }
    for (int o = 16; o; o >>= 1) s += __shfl_xor(s, o, 32);
    float inv = 1.f / s;
    for (int i = 0; i < nv; ++i) pr[lane + (i << 5)] = (__bf16)(v[i] * inv);
}

// ------------------------------- host side ----------------------------------
static void run_gemm(hipStream_t st,
    const __bf16* A, long long lda, long long sa0, long long sa1, long long sa2,
    const __bf16* B, long long ldb, long long sb0, long long sb1, long long sb2, int opB,
    float* C, __bf16* Cbf, long long ldc, long long sc0, long long sc1, long long sc2,
    const float* Cin, const float* bias, const float* bias2, const int* rowmap, int b2ld,
    int M, int N, int K, int batch, int D1, int D2, int act)
{
    GP p;
    p.A = A; p.lda = lda; p.sa0 = sa0; p.sa1 = sa1; p.sa2 = sa2;
    p.B = B; p.ldb = ldb; p.sb0 = sb0; p.sb1 = sb1; p.sb2 = sb2;
    p.C = C; p.ldc = ldc; p.sc0 = sc0; p.sc1 = sc1; p.sc2 = sc2;
    p.Cbf = Cbf; p.Cin = Cin; p.bias = bias; p.bias2 = bias2; p.rowmap = rowmap;
    p.b2ld = b2ld; p.M = M; p.N = N; p.K = K; p.D1 = D1; p.D2 = D2; p.opB = opB; p.act = act;
    dim3 grid((N + GBN - 1) / GBN, (M + GBM - 1) / GBM, batch);
    k_gemm<<<grid, dim3(256), 0, st>>>(p);
}

extern "C" void kernel_launch(void* const* d_in, const int* in_sizes, int n_in,
                              void* d_out, int out_size, void* d_ws, size_t ws_size,
                              hipStream_t stream) {
    // dims
    const int Bd = 8, T = 128, Ld = 100, TR = 32, HIDc = 128, Hh = 3, HD = 384, S2 = 32, FF = 512;
    const long long NTOK = (long long)Bd * T * Ld;   // 102400
    const long long NKV  = (long long)Bd * Ld * S2;  // 25600
    const int NBT = Bd * T;                          // 1024
    const float SCALE = 1.0f / sqrtf((float)HIDc);

    const float* in_x  = (const float*)d_in[0];
    const float* in_nx = (const float*)d_in[1];
    const int*   in_t  = (const int*)d_in[2];
    // params flattened in recursive alphabetical (pytree) key order after the 3 arrays
    auto PL = [&](int i) -> const float* { return (const float*)d_in[3 + i]; };
    // per-block leaf layout (47 leaves): be1..be5(+0..4), comm(+5..15), g1..g5(+16..20),
    // mlp(+21..24: W1,W2,b1,b2), spat(+25..35), temp(+36..46)
    // attn leaf order: Wk,Wkt,Wo,Wq,Wqt,Wv,Wvt,bkt,bo,bqt,bvt
    const int PB_DIFF = 141, PB_NX = 145, PB_OUT = 149, PB_STEP = 153, PB_X = 157;

    // deterministic bump allocator over d_ws (256B align + 64KB guard per alloc)
    char* wsb = (char*)d_ws;
    size_t cur = 0;
    auto alloc = [&](size_t bytes) -> void* {
        void* p = wsb + cur;
        cur += bytes;
        cur = (cur + 255) & ~(size_t)255;
        cur += 65536;
        return p;
    };
    auto g1d = [](long long n) { return dim3((unsigned)((n + 255) / 256)); };
    auto cvt = [&](const float* src, long long n) -> __bf16* {
        __bf16* d = (__bf16*)alloc((size_t)n * 2);
        k_cvt<<<g1d(n), 256, 0, stream>>>(src, d, n);
        return d;
    };

    // ---- persistent buffers ----
    int*    rmA   = (int*)alloc(NTOK * 4);
    int*    rmB   = (int*)alloc(NKV * 4);
    __bf16* SINB  = (__bf16*)alloc((size_t)8 * 128 * 2);
    __bf16* SINB2 = (__bf16*)alloc((size_t)128 * 128 * 2);
    float*  DTE   = (float*)alloc((size_t)8 * 128 * 4);
    float*  STE   = (float*)alloc((size_t)128 * 128 * 4);
    __bf16* TEM   = (__bf16*)alloc((size_t)NBT * 128 * 2);
    float*  TEQ   = (float*)alloc((size_t)NBT * HD * 4);
    float*  TEK   = (float*)alloc((size_t)NBT * HD * 4);
    float*  TEV   = (float*)alloc((size_t)NBT * HD * 4);
    float*  hA    = (float*)alloc((size_t)NTOK * 128 * 4);
    float*  hB    = (float*)alloc((size_t)NTOK * 128 * 4);
    float*  HNF   = (float*)alloc((size_t)NKV * 128 * 4);
    __bf16* HNBF  = (__bf16*)alloc((size_t)NKV * 128 * 2);
    __bf16* GNBF  = (__bf16*)alloc((size_t)NTOK * 128 * 2);
    __bf16* XBF   = (__bf16*)alloc((size_t)NTOK * TR * 2);
    __bf16* NXBF  = (__bf16*)alloc((size_t)NKV * TR * 2);
    __bf16* H1    = (__bf16*)alloc((size_t)NTOK * FF * 2);
    __bf16* Qb    = (__bf16*)alloc((size_t)NTOK * HD * 2);
    __bf16* Kb    = (__bf16*)alloc((size_t)NTOK * HD * 2);
    __bf16* Vb    = (__bf16*)alloc((size_t)NTOK * HD * 2);
    long long SCmax = (long long)3072 * 100 * 128;   // == 2400*128*128
    float*  SC    = (float*)alloc((size_t)SCmax * 4);
    __bf16* PR    = (__bf16*)alloc((size_t)SCmax * 2);
    __bf16* AO    = (__bf16*)alloc((size_t)NTOK * HD * 2);
    float*  TF    = (float*)alloc((size_t)NTOK * 128 * 4);
    float*  TF2   = (float*)alloc((size_t)NTOK * 128 * 4);
    float*  STAT  = (float*)alloc((size_t)128 * 4);

    // ---- convert weights to bf16 ----
    struct AttnB { const __bf16 *Wk,*Wkt,*Wo,*Wq,*Wqt,*Wv,*Wvt; const float *bkt,*bo,*bqt,*bvt; };
    AttnB ab[3][3];                    // [block][0=comm,1=spat,2=temp]
    const __bf16 *mW1[3], *mW2[3];
    const float  *mb1[3], *mb2[3];
    const float  *ga[3][5], *be[3][5];
    for (int b = 0; b < 3; ++b) {
        int base = b * 47;
        int mbase[3] = { base + 5, base + 25, base + 36 };
        for (int m = 0; m < 3; ++m) {
            int p0 = mbase[m];
            ab[b][m].Wk  = cvt(PL(p0 + 0), 49152);
            ab[b][m].Wkt = cvt(PL(p0 + 1), 49152);
            ab[b][m].Wo  = cvt(PL(p0 + 2), 49152);
            ab[b][m].Wq  = cvt(PL(p0 + 3), 49152);
            ab[b][m].Wqt = cvt(PL(p0 + 4), 49152);
            ab[b][m].Wv  = cvt(PL(p0 + 5), 49152);
            ab[b][m].Wvt = cvt(PL(p0 + 6), 49152);
            ab[b][m].bkt = PL(p0 + 7); ab[b][m].bo = PL(p0 + 8);
            ab[b][m].bqt = PL(p0 + 9); ab[b][m].bvt = PL(p0 + 10);
        }
        mW1[b] = cvt(PL(base + 21), 65536);
        mW2[b] = cvt(PL(base + 22), 65536);
        mb1[b] = PL(base + 23); mb2[b] = PL(base + 24);
        for (int i = 0; i < 5; ++i) { be[b][i] = PL(base + i); ga[b][i] = PL(base + 16 + i); }
    }
    const __bf16* dW1 = cvt(PL(PB_DIFF + 0), 65536); const __bf16* dW2 = cvt(PL(PB_DIFF + 1), 65536);
    const float *db1 = PL(PB_DIFF + 2), *db2 = PL(PB_DIFF + 3);
    const __bf16* nW1 = cvt(PL(PB_NX + 0), 16384);   const __bf16* nW2 = cvt(PL(PB_NX + 1), 65536);
    const float *nb1 = PL(PB_NX + 2), *nb2 = PL(PB_NX + 3);
    const __bf16* oW1 = cvt(PL(PB_OUT + 0), 65536);  const __bf16* oW2 = cvt(PL(PB_OUT + 1), 16384);
    const float *ob1 = PL(PB_OUT + 2), *ob2 = PL(PB_OUT + 3);
    const __bf16* sW1 = cvt(PL(PB_STEP + 0), 65536); const __bf16* sW2 = cvt(PL(PB_STEP + 1), 65536);
    const float *sb1 = PL(PB_STEP + 2), *sb2 = PL(PB_STEP + 3);
    const __bf16* xW1 = cvt(PL(PB_X + 0), 16384);    const __bf16* xW2 = cvt(PL(PB_X + 1), 65536);
    const float *xb1 = PL(PB_X + 2), *xb2 = PL(PB_X + 3);

    // ---- rowmaps: token row -> time-table row ----
    k_rowmap<<<g1d(NTOK), 256, 0, stream>>>(rmA, (int)NTOK, Ld, 1, 1, 1);          // b*T+t = r/L
    k_rowmap<<<g1d(NKV), 256, 0, stream>>>(rmB, (int)NKV, Ld * S2, T, 4, 8);       // b*T + t2

    // ---- time embeddings: dte = mlp2(sinemb(time)); ste = mlp2(sinemb(arange)) ----
    k_sinemb<<<g1d(8 * 64), 256, 0, stream>>>(in_t, 1, 8, SINB);
    run_gemm(stream, SINB, 128, 0,0,0, dW1, FF, 0,0,0, 0, nullptr, H1, FF, 0,0,0,
             nullptr, db1, nullptr, nullptr, 0, 8, FF, 128, 1,1,1, 1);
    run_gemm(stream, H1, FF, 0,0,0, dW2, 128, 0,0,0, 0, DTE, nullptr, 128, 0,0,0,
             nullptr, db2, nullptr, nullptr, 0, 8, 128, FF, 1,1,1, 0);
    k_sinemb<<<g1d(128 * 64), 256, 0, stream>>>(nullptr, 0, 128, SINB2);
    run_gemm(stream, SINB2, 128, 0,0,0, sW1, FF, 0,0,0, 0, nullptr, H1, FF, 0,0,0,
             nullptr, sb1, nullptr, nullptr, 0, 128, FF, 128, 1,1,1, 1);
    run_gemm(stream, H1, FF, 0,0,0, sW2, 128, 0,0,0, 0, STE, nullptr, 128, 0,0,0,
             nullptr, sb2, nullptr, nullptr, 0, 128, 128, FF, 1,1,1, 0);
    k_temish<<<g1d(NBT * 128), 256, 0, stream>>>(DTE, STE, TEM);

    // ---- input embeddings: h = mlp2(x); hn = mlp2(neighbor_x) ----
    k_cvt<<<g1d(NTOK * TR), 256, 0, stream>>>(in_x, XBF, NTOK * TR);
    run_gemm(stream, XBF, TR, 0,0,0, xW1, FF, 0,0,0, 0, nullptr, H1, FF, 0,0,0,
             nullptr, xb1, nullptr, nullptr, 0, (int)NTOK, FF, TR, 1,1,1, 1);
    run_gemm(stream, H1, FF, 0,0,0, xW2, 128, 0,0,0, 0, hA, nullptr, 128, 0,0,0,
             nullptr, xb2, nullptr, nullptr, 0, (int)NTOK, 128, FF, 1,1,1, 0);
    k_cvt<<<g1d(NKV * TR), 256, 0, stream>>>(in_nx, NXBF, NKV * TR);
    run_gemm(stream, NXBF, TR, 0,0,0, nW1, FF, 0,0,0, 0, nullptr, H1, FF, 0,0,0,
             nullptr, nb1, nullptr, nullptr, 0, (int)NKV, FF, TR, 1,1,1, 1);
    run_gemm(stream, H1, FF, 0,0,0, nW2, 128, 0,0,0, 0, HNF, nullptr, 128, 0,0,0,
             nullptr, nb2, nullptr, nullptr, 0, (int)NKV, 128, FF, 1,1,1, 0);

    float* hCur = hA;
    float* hNxt = hB;
    const long long TK = (long long)T * Ld * HD;   // token slab stride (b)
    const long long LK = (long long)Ld * HD;       // per-t slab (spat) / row stride (temp)
    const long long KVK = (long long)Ld * S2 * HD; // kv batch stride (b)

    auto te_tables = [&](const AttnB& w) {
        run_gemm(stream, TEM, 128, 0,0,0, w.Wqt, HD, 0,0,0, 0, TEQ, nullptr, HD, 0,0,0,
                 nullptr, w.bqt, nullptr, nullptr, 0, NBT, HD, 128, 1,1,1, 0);
        run_gemm(stream, TEM, 128, 0,0,0, w.Wkt, HD, 0,0,0, 0, TEK, nullptr, HD, 0,0,0,
                 nullptr, w.bkt, nullptr, nullptr, 0, NBT, HD, 128, 1,1,1, 0);
        run_gemm(stream, TEM, 128, 0,0,0, w.Wvt, HD, 0,0,0, 0, TEV, nullptr, HD, 0,0,0,
                 nullptr, w.bvt, nullptr, nullptr, 0, NBT, HD, 128, 1,1,1, 0);
    };
    auto gnorm = [&](const float* x, int Rows, const float* g, const float* bta,
                     float* of, __bf16* ob) {
        k_gn_stats<<<dim3(Bd * 8), 256, 0, stream>>>(x, STAT, Rows);
        long long n = (long long)Bd * Rows * 128;
        k_gn_apply<<<g1d(n), 256, 0, stream>>>(x, STAT, g, bta, of, ob, Rows, n);
    };
    auto smrows = [](long long rows) { return dim3((unsigned)((rows + 7) / 8)); };

    for (int b = 0; b < 3; ++b) {
        // ================= comm attention =================
        gnorm(hCur, T * Ld, ga[b][0], be[b][0], nullptr, GNBF);
        gnorm(HNF, Ld * S2, ga[b][1], be[b][1], nullptr, HNBF);
        te_tables(ab[b][0]);
        run_gemm(stream, GNBF, 128, 0,0,0, ab[b][0].Wq, HD, 0,0,0, 0, nullptr, Qb, HD, 0,0,0,
                 nullptr, nullptr, TEQ, rmA, HD, (int)NTOK, HD, 128, 1,1,1, 0);
        run_gemm(stream, HNBF, 128, 0,0,0, ab[b][0].Wk, HD, 0,0,0, 0, nullptr, Kb, HD, 0,0,0,
                 nullptr, nullptr, TEK, rmB, HD, (int)NKV, HD, 128, 1,1,1, 0);
        run_gemm(stream, HNBF, 128, 0,0,0, ab[b][0].Wv, HD, 0,0,0, 0, nullptr, Vb, HD, 0,0,0,
                 nullptr, nullptr, TEV, rmB, HD, (int)NKV, HD, 128, 1,1,1, 0);
        // scores (b,l,h): T x S2, K=128
        run_gemm(stream, Qb, LK, TK, HD, 128,
                 Kb, HD, KVK, (long long)S2 * HD, 128, 1,
                 SC, nullptr, S2, (long long)Ld * Hh * T * S2, (long long)Hh * T * S2, (long long)T * S2,
                 nullptr, nullptr, nullptr, nullptr, 0,
                 T, S2, 128, Bd * Ld * Hh, Ld, Hh, 0);
        k_softmax<<<smrows((long long)Bd * Ld * Hh * T), 256, 0, stream>>>(
            SC, PR, (long long)Bd * Ld * Hh * T, S2, S2, SCALE);
        // AV (b,l,h): T x 128, K=32; scatter to (b,t,l,h,d)
        run_gemm(stream, PR, S2, (long long)Ld * Hh * T * S2, (long long)Hh * T * S2, (long long)T * S2,
                 Vb, HD, KVK, (long long)S2 * HD, 128, 0,
                 nullptr, AO, LK, TK, HD, 128,
                 nullptr, nullptr, nullptr, nullptr, 0,
                 T, 128, S2, Bd * Ld * Hh, Ld, Hh, 0);
        run_gemm(stream, AO, HD, 0,0,0, ab[b][0].Wo, 128, 0,0,0, 0, hNxt, nullptr, 128, 0,0,0,
                 hCur, ab[b][0].bo, nullptr, nullptr, 0, (int)NTOK, 128, HD, 1,1,1, 0);
        { float* t = hCur; hCur = hNxt; hNxt = t; }

        // ================= spatial attention =================
        gnorm(hCur, T * Ld, ga[b][2], be[b][2], nullptr, GNBF);
        te_tables(ab[b][1]);
        run_gemm(stream, GNBF, 128, 0,0,0, ab[b][1].Wq, HD, 0,0,0, 0, nullptr, Qb, HD, 0,0,0,
                 nullptr, nullptr, TEQ, rmA, HD, (int)NTOK, HD, 128, 1,1,1, 0);
        run_gemm(stream, GNBF, 128, 0,0,0, ab[b][1].Wk, HD, 0,0,0, 0, nullptr, Kb, HD, 0,0,0,
                 nullptr, nullptr, TEK, rmA, HD, (int)NTOK, HD, 128, 1,1,1, 0);
        run_gemm(stream, GNBF, 128, 0,0,0, ab[b][1].Wv, HD, 0,0,0, 0, nullptr, Vb, HD, 0,0,0,
                 nullptr, nullptr, TEV, rmA, HD, (int)NTOK, HD, 128, 1,1,1, 0);
        // scores (b,t,h): L x L (pad cols to 128)
        run_gemm(stream, Qb, HD, TK, LK, 128,
                 Kb, HD, TK, LK, 128, 1,
                 SC, nullptr, 128, (long long)T * Hh * Ld * 128, (long long)Hh * Ld * 128, (long long)Ld * 128,
                 nullptr, nullptr, nullptr, nullptr, 0,
                 Ld, Ld, 128, Bd * T * Hh, T, Hh, 0);
        k_softmax<<<smrows((long long)Bd * T * Hh * Ld), 256, 0, stream>>>(
            SC, PR, (long long)Bd * T * Hh * Ld, Ld, 128, SCALE);
        // AV (b,t,h): L x 128, K=128 (padded; pad probs are zero)
        run_gemm(stream, PR, 128, (long long)T * Hh * Ld * 128, (long long)Hh * Ld * 128, (long long)Ld * 128,
                 Vb, HD, TK, LK, 128, 0,
                 nullptr, AO, HD, TK, LK, 128,
                 nullptr, nullptr, nullptr, nullptr, 0,
                 Ld, 128, 128, Bd * T * Hh, T, Hh, 0);
        run_gemm(stream, AO, HD, 0,0,0, ab[b][1].Wo, 128, 0,0,0, 0, TF, nullptr, 128, 0,0,0,
                 nullptr, ab[b][1].bo, nullptr, nullptr, 0, (int)NTOK, 128, HD, 1,1,1, 0);

        // ================= temporal attention =================
        gnorm(hCur, T * Ld, ga[b][3], be[b][3], nullptr, GNBF);
        te_tables(ab[b][2]);
        run_gemm(stream, GNBF, 128, 0,0,0, ab[b][2].Wq, HD, 0,0,0, 0, nullptr, Qb, HD, 0,0,0,
                 nullptr, nullptr, TEQ, rmA, HD, (int)NTOK, HD, 128, 1,1,1, 0);
        run_gemm(stream, GNBF, 128, 0,0,0, ab[b][2].Wk, HD, 0,0,0, 0, nullptr, Kb, HD, 0,0,0,
                 nullptr, nullptr, TEK, rmA, HD, (int)NTOK, HD, 128, 1,1,1, 0);
        run_gemm(stream, GNBF, 128, 0,0,0, ab[b][2].Wv, HD, 0,0,0, 0, nullptr, Vb, HD, 0,0,0,
                 nullptr, nullptr, TEV, rmA, HD, (int)NTOK, HD, 128, 1,1,1, 0);
        // scores (b,l,h): T x T, rows strided by L*HD
        run_gemm(stream, Qb, LK, TK, HD, 128,
                 Kb, LK, TK, HD, 128, 1,
                 SC, nullptr, 128, (long long)Ld * Hh * T * 128, (long long)Hh * T * 128, (long long)T * 128,
                 nullptr, nullptr, nullptr, nullptr, 0,
                 T, T, 128, Bd * Ld * Hh, Ld, Hh, 0);
        k_softmax<<<smrows((long long)Bd * Ld * Hh * T), 256, 0, stream>>>(
            SC, PR, (long long)Bd * Ld * Hh * T, T, 128, SCALE);
        run_gemm(stream, PR, 128, (long long)Ld * Hh * T * 128, (long long)Hh * T * 128, (long long)T * 128,
                 Vb, LK, TK, HD, 128, 0,
                 nullptr, AO, LK, TK, HD, 128,
                 nullptr, nullptr, nullptr, nullptr, 0,
                 T, 128, T, Bd * Ld * Hh, Ld, Hh, 0);
        run_gemm(stream, AO, HD, 0,0,0, ab[b][2].Wo, 128, 0,0,0, 0, TF2, nullptr, 128, 0,0,0,
                 TF, ab[b][2].bo, nullptr, nullptr, 0, (int)NTOK, 128, HD, 1,1,1, 0);
        k_add2<<<g1d(NTOK * 128), 256, 0, stream>>>(hCur, TF2, hNxt, NTOK * 128);
        { float* t = hCur; hCur = hNxt; hNxt = t; }

        // ================= gnorm5 + MLP =================
        gnorm(hCur, T * Ld, ga[b][4], be[b][4], TF, GNBF);
        run_gemm(stream, GNBF, 128, 0,0,0, mW1[b], FF, 0,0,0, 0, nullptr, H1, FF, 0,0,0,
                 nullptr, mb1[b], nullptr, nullptr, 0, (int)NTOK, FF, 128, 1,1,1, 1);
        run_gemm(stream, H1, FF, 0,0,0, mW2[b], 128, 0,0,0, 0, hNxt, nullptr, 128, 0,0,0,
                 TF, mb2[b], nullptr, nullptr, 0, (int)NTOK, 128, FF, 1,1,1, 0);
        { float* t = hCur; hCur = hNxt; hNxt = t; }
    }

    // ---- output head ----
    k_cvt<<<g1d(NTOK * 128), 256, 0, stream>>>(hCur, GNBF, NTOK * 128);
    run_gemm(stream, GNBF, 128, 0,0,0, oW1, FF, 0,0,0, 0, nullptr, H1, FF, 0,0,0,
             nullptr, ob1, nullptr, nullptr, 0, (int)NTOK, FF, 128, 1,1,1, 1);
    run_gemm(stream, H1, FF, 0,0,0, oW2, TR, 0,0,0, 0, (float*)d_out, nullptr, TR, 0,0,0,
             nullptr, ob2, nullptr, nullptr, 0, (int)NTOK, TR, FF, 1,1,1, 0);

    (void)in_sizes; (void)n_in; (void)out_size; (void)ws_size;
}
